// DictionaryLabelAttention_16561393893641
// MI455X (gfx1250) — matrix operand order repaired and checked by reading, not compiled, not run on hardware
//
#include <hip/hip_runtime.h>
#include <hip/hip_bf16.h>
#include <stdint.h>

// ---------------------------------------------------------------------------
// CDNA5 (gfx1250) wave32 WMMA implementation of the dictionary-label-attention
// reference.  All heavy GEMMs run on v_wmma_f32_16x16x32_bf16 with fp32 accum.
// Attention kernel uses 2x2 register blocking to halve L2/LDS bytes per WMMA.
// ---------------------------------------------------------------------------

typedef __attribute__((ext_vector_type(16))) __bf16 v16bf;
typedef __attribute__((ext_vector_type(8)))  __bf16 v8bf;
typedef __attribute__((ext_vector_type(8)))  float  v8f;

#define BB   4
#define SS   4096
#define DD   768
#define MM   3072
#define CC   8921
#define CPAD 8928            // 558 * 16
#define ROWS (BB * SS)       // 16384

#define KT_ENC (DD / 32)     // 24
#define KT_DEC (MM / 32)     // 96
#define NT_ENC (MM / 16)     // 192
#define NT_DEC (DD / 16)     // 48
#define NT_C   (CPAD / 16)   // 558

static __device__ __forceinline__ unsigned short f2bf(float f) {
  union { float f; unsigned u; } v; v.f = f;
  unsigned u = v.u;
  u += 0x7FFFu + ((u >> 16) & 1u);   // round-to-nearest-even to bf16
  return (unsigned short)(u >> 16);
}

static __device__ __forceinline__ v8f wmma_bf16(v16bf a, v16bf b, v8f c) {
  // D = A(16x32 bf16) * B(32x16 bf16) + C(16x16 f32)
  return __builtin_amdgcn_wmma_f32_16x16x32_bf16(false, a, false, b,
                                                 (short)0, c, false, false);
}

union AFrag { v16bf v; v8bf h[2]; };

// Per-lane A fragment from row-major bf16 (ushort) storage.
// ISA 16-bit A 16x32 layout: lanes 0-15 -> K 0..7 & 16..23; lanes 16-31 -> K 8..15 & 24..31.
static __device__ __forceinline__ v16bf load_a(const unsigned short* rowbase, int k0) {
  AFrag a;
  a.h[0] = *(const v8bf*)(rowbase + k0);
  a.h[1] = *(const v8bf*)(rowbase + k0 + 16);
  return a.v;
}

// ---------------------------------------------------------------------------
__global__ void k_f32_to_bf16(const float* __restrict__ src,
                              unsigned short* __restrict__ dst, int n) {
  int i = blockIdx.x * blockDim.x + threadIdx.x;
  int stride = gridDim.x * blockDim.x;
  for (; i < n; i += stride) dst[i] = f2bf(src[i]);
}

// Pack a (K x N) operand, element (k,n) at src[k*sk + n*sn], into WMMA-B tile
// layout: tile = nb*Kt + kb holds 512 bf16; lane owns n = nb*16 + (lane&15),
// k = kb*32 + 8*(lane>>4) + e + (e>=8 ? 8 : 0)  (e = 0..15, 32 contiguous bytes per lane).
__global__ void k_pack_b(const float* __restrict__ src,
                         unsigned short* __restrict__ dst,
                         int Kt, int Nt, int Nvalid, long sk, long sn) {
  int tid = blockIdx.x * blockDim.x + threadIdx.x;
  int total = Kt * Nt * 32;
  if (tid >= total) return;
  int lane = tid & 31;
  int tile = tid >> 5;
  int kb = tile % Kt;
  int nb = tile / Kt;
  int n  = nb * 16 + (lane & 15);
  int k0 = kb * 32 + ((lane >> 4) * 8);
  unsigned short* out = dst + (size_t)tile * 512 + lane * 16;
  #pragma unroll
  for (int e = 0; e < 16; ++e) {
    float v = 0.0f;
    if (n < Nvalid) v = src[(long)(k0 + e + (e >= 8 ? 8 : 0)) * sk + (long)n * sn];
    out[e] = f2bf(v);
  }
}

// ---------------------------------------------------------------------------
// f = relu(x @ w_enc + b_enc), stored bf16; also accumulates sum(|f|).
// Each wave computes TWO 16x16 output tiles (nb0, nb0+96) sharing A fragments.
__global__ __launch_bounds__(256) void k_encode(
    const unsigned short* __restrict__ Xb, const unsigned short* __restrict__ Wp,
    const float* __restrict__ b_enc, unsigned short* __restrict__ F,
    float* __restrict__ acc) {
  const int lane = threadIdx.x & 31;
  const int wave = threadIdx.x >> 5;
  const int m0 = blockIdx.x * 16;
  const int nb0 = blockIdx.y * 8 + wave;          // 0..95
  const int nb1 = nb0 + NT_ENC / 2;               // 96..191
  const unsigned short* arow = Xb + (size_t)(m0 + (lane & 15)) * DD + ((lane >> 4) * 8);
  const unsigned short* bt0 = Wp + (size_t)nb0 * KT_ENC * 512 + lane * 16;
  const unsigned short* bt1 = Wp + (size_t)nb1 * KT_ENC * 512 + lane * 16;
  v8f c0 = {0.f,0.f,0.f,0.f,0.f,0.f,0.f,0.f};
  v8f c1 = {0.f,0.f,0.f,0.f,0.f,0.f,0.f,0.f};
  #pragma unroll 4
  for (int kb = 0; kb < KT_ENC; ++kb) {
    v16bf a = load_a(arow, kb * 32);
    v16bf b0 = *(const v16bf*)(bt0 + (size_t)kb * 512);
    v16bf b1 = *(const v16bf*)(bt1 + (size_t)kb * 512);
    c0 = wmma_bf16(a, b0, c0);
    c1 = wmma_bf16(a, b1, c1);
  }
  const int col = lane & 15;
  const int rb  = (lane >> 4) * 8;   // C/D layout: lanes 16-31 hold M = 8..15
  float asum = 0.f;
  const float bias0 = b_enc[nb0 * 16 + col];
  const float bias1 = b_enc[nb1 * 16 + col];
  #pragma unroll
  for (int i = 0; i < 8; ++i) {
    float v0 = c0[i] + bias0; v0 = v0 > 0.f ? v0 : 0.f;
    float v1 = c1[i] + bias1; v1 = v1 > 0.f ? v1 : 0.f;
    asum += v0 + v1;
    F[(size_t)(m0 + rb + i) * MM + nb0 * 16 + col] = f2bf(v0);
    F[(size_t)(m0 + rb + i) * MM + nb1 * 16 + col] = f2bf(v1);
  }
  asum += __shfl_xor(asum, 16); asum += __shfl_xor(asum, 8);
  asum += __shfl_xor(asum, 4);  asum += __shfl_xor(asum, 2);
  asum += __shfl_xor(asum, 1);
  if (lane == 0) atomicAdd(&acc[1], asum);
}

// x_hat = f @ w_dec + b_dec; accumulates sum((x_hat - x)^2); x_hat not stored.
__global__ __launch_bounds__(256) void k_decode(
    const unsigned short* __restrict__ F, const unsigned short* __restrict__ Wp,
    const float* __restrict__ b_dec, const float* __restrict__ x,
    float* __restrict__ acc) {
  const int lane = threadIdx.x & 31;
  const int wave = threadIdx.x >> 5;
  const int m0 = blockIdx.x * 16;
  const int nb0 = blockIdx.y * 8 + wave;          // 0..23
  const int nb1 = nb0 + NT_DEC / 2;               // 24..47
  const unsigned short* arow = F + (size_t)(m0 + (lane & 15)) * MM + ((lane >> 4) * 8);
  const unsigned short* bt0 = Wp + (size_t)nb0 * KT_DEC * 512 + lane * 16;
  const unsigned short* bt1 = Wp + (size_t)nb1 * KT_DEC * 512 + lane * 16;
  v8f c0 = {0.f,0.f,0.f,0.f,0.f,0.f,0.f,0.f};
  v8f c1 = {0.f,0.f,0.f,0.f,0.f,0.f,0.f,0.f};
  #pragma unroll 4
  for (int kb = 0; kb < KT_DEC; ++kb) {
    v16bf a = load_a(arow, kb * 32);
    v16bf b0 = *(const v16bf*)(bt0 + (size_t)kb * 512);
    v16bf b1 = *(const v16bf*)(bt1 + (size_t)kb * 512);
    c0 = wmma_bf16(a, b0, c0);
    c1 = wmma_bf16(a, b1, c1);
  }
  const int col = lane & 15;
  const int rb  = (lane >> 4) * 8;
  const float bias0 = b_dec[nb0 * 16 + col];
  const float bias1 = b_dec[nb1 * 16 + col];
  float ss = 0.f;
  #pragma unroll
  for (int i = 0; i < 8; ++i) {
    float xh0 = c0[i] + bias0;
    float d0 = xh0 - x[(size_t)(m0 + rb + i) * DD + nb0 * 16 + col];
    float xh1 = c1[i] + bias1;
    float d1 = xh1 - x[(size_t)(m0 + rb + i) * DD + nb1 * 16 + col];
    ss += d0 * d0 + d1 * d1;
  }
  ss += __shfl_xor(ss, 16); ss += __shfl_xor(ss, 8);
  ss += __shfl_xor(ss, 4);  ss += __shfl_xor(ss, 2);
  ss += __shfl_xor(ss, 1);
  if (lane == 0) atomicAdd(&acc[0], ss);
}

// ---------------------------------------------------------------------------
// Fused label attention (2x2 register-blocked):
//   attn[s,c] = f[b,s,:] . icd[c,:]      (WMMA, K = 3072)
//   xw[s,c]   = x[b,s,:] . out_w[c,:]    (WMMA, K = 768)
//   logits[b,c] = sum_s softmax_s(attn) * xw + out_b[c]   (online softmax)
// One workgroup per (32-label block, batch).  Packed B operands for both label
// tiles staged once in LDS (240 KB <= 320 KB WGP budget).  Each wave processes
// 2 s-chunks x 2 label tiles per iteration: 2 A frags + 2 B frags -> 4 WMMAs,
// halving both L2-A and LDS-B bytes per WMMA vs. the unblocked version.
__global__ __launch_bounds__(256) void k_label_attn(
    const unsigned short* __restrict__ F, const unsigned short* __restrict__ Xb,
    const unsigned short* __restrict__ ICDp, const unsigned short* __restrict__ OWp,
    const float* __restrict__ out_b, float* __restrict__ logits) {
  extern __shared__ unsigned short smem[];
  unsigned short* lds_icd = smem;                              // 2*96 tiles = 192 KB
  unsigned short* lds_ow  = smem + 2 * KT_DEC * 512;           // 2*24 tiles =  48 KB
  float* red = (float*)(smem + 2 * (KT_DEC + KT_ENC) * 512);   // 8*32*3 floats

  const int lane = threadIdx.x & 31;
  const int wave = threadIdx.x >> 5;
  const int nb0 = blockIdx.x * 2;    // first of two label tiles
  const int b   = blockIdx.y;        // batch

  // Cooperative LDS fill: the two label tiles are contiguous in packed layout.
  {
    const uint4* gi = (const uint4*)(ICDp + (size_t)nb0 * KT_DEC * 512);
    uint4* li = (uint4*)lds_icd;
    for (int i = threadIdx.x; i < 2 * KT_DEC * 64; i += 256) li[i] = gi[i];
    const uint4* go = (const uint4*)(OWp + (size_t)nb0 * KT_ENC * 512);
    uint4* lo = (uint4*)lds_ow;
    for (int i = threadIdx.x; i < 2 * KT_ENC * 64; i += 256) lo[i] = go[i];
  }
  __syncthreads();

  // Per-lane online softmax state, one set per label subtile.
  float m0s = -3.0e38f, Z0 = 0.f, N0 = 0.f;
  float m1s = -3.0e38f, Z1 = 0.f, N1 = 0.f;

  const unsigned short* ic0 = lds_icd + lane * 16;
  const unsigned short* ic1 = lds_icd + (size_t)KT_DEC * 512 + lane * 16;
  const unsigned short* ow0 = lds_ow + lane * 16;
  const unsigned short* ow1 = lds_ow + (size_t)KT_ENC * 512 + lane * 16;

  for (int p = wave; p < SS / 32; p += 8) {       // 2 adjacent s-chunks per iter
    const int row0 = b * SS + p * 32 + (lane & 15);
    const unsigned short* frow0 = F  + (size_t)row0 * MM + ((lane >> 4) * 8);
    const unsigned short* frow1 = frow0 + (size_t)16 * MM;
    const unsigned short* xrow0 = Xb + (size_t)row0 * DD + ((lane >> 4) * 8);
    const unsigned short* xrow1 = xrow0 + (size_t)16 * DD;

    v8f at00 = {0.f,0.f,0.f,0.f,0.f,0.f,0.f,0.f};
    v8f at01 = at00, at10 = at00, at11 = at00;
    v8f wt00 = at00, wt01 = at00, wt10 = at00, wt11 = at00;

    #pragma unroll 2
    for (int kb = 0; kb < KT_DEC; ++kb) {
      v16bf a0 = load_a(frow0, kb * 32);
      v16bf a1 = load_a(frow1, kb * 32);
      v16bf b0 = *(const v16bf*)(ic0 + (size_t)kb * 512);
      v16bf b1 = *(const v16bf*)(ic1 + (size_t)kb * 512);
      at00 = wmma_bf16(a0, b0, at00);
      at01 = wmma_bf16(a0, b1, at01);
      at10 = wmma_bf16(a1, b0, at10);
      at11 = wmma_bf16(a1, b1, at11);
    }
    #pragma unroll 2
    for (int kb = 0; kb < KT_ENC; ++kb) {
      v16bf a0 = load_a(xrow0, kb * 32);
      v16bf a1 = load_a(xrow1, kb * 32);
      v16bf b0 = *(const v16bf*)(ow0 + (size_t)kb * 512);
      v16bf b1 = *(const v16bf*)(ow1 + (size_t)kb * 512);
      wt00 = wmma_bf16(a0, b0, wt00);
      wt01 = wmma_bf16(a0, b1, wt01);
      wt10 = wmma_bf16(a1, b0, wt10);
      wt11 = wmma_bf16(a1, b1, wt11);
    }
    // Fold 8 (s,c) entries per accumulator into the per-label online state.
    #pragma unroll
    for (int i = 0; i < 8; ++i) {
      { float a = at00[i], w = wt00[i];
        float nm = fmaxf(m0s, a); float e0 = __expf(m0s - nm); float pr = __expf(a - nm);
        Z0 = Z0 * e0 + pr; N0 = N0 * e0 + pr * w; m0s = nm; }
      { float a = at10[i], w = wt10[i];
        float nm = fmaxf(m0s, a); float e0 = __expf(m0s - nm); float pr = __expf(a - nm);
        Z0 = Z0 * e0 + pr; N0 = N0 * e0 + pr * w; m0s = nm; }
      { float a = at01[i], w = wt01[i];
        float nm = fmaxf(m1s, a); float e0 = __expf(m1s - nm); float pr = __expf(a - nm);
        Z1 = Z1 * e0 + pr; N1 = N1 * e0 + pr * w; m1s = nm; }
      { float a = at11[i], w = wt11[i];
        float nm = fmaxf(m1s, a); float e0 = __expf(m1s - nm); float pr = __expf(a - nm);
        Z1 = Z1 * e0 + pr; N1 = N1 * e0 + pr * w; m1s = nm; }
    }
  }
  // Combine lane L (s rows 0..7) with lane L+16 (s rows 8..15): same label.
  {
    float m2 = __shfl_xor(m0s, 16), Z2 = __shfl_xor(Z0, 16), N2 = __shfl_xor(N0, 16);
    float nm = fmaxf(m0s, m2);
    float e1 = __expf(m0s - nm), e2 = __expf(m2 - nm);
    Z0 = Z0 * e1 + Z2 * e2; N0 = N0 * e1 + N2 * e2; m0s = nm;
  }
  {
    float m2 = __shfl_xor(m1s, 16), Z2 = __shfl_xor(Z1, 16), N2 = __shfl_xor(N1, 16);
    float nm = fmaxf(m1s, m2);
    float e1 = __expf(m1s - nm), e2 = __expf(m2 - nm);
    Z1 = Z1 * e1 + Z2 * e2; N1 = N1 * e1 + N2 * e2; m1s = nm;
  }
  if (lane < 16) {
    float* r0 = red + (wave * 32 + lane) * 3;        // subtile 0
    r0[0] = m0s; r0[1] = Z0; r0[2] = N0;
    float* r1 = red + (wave * 32 + 16 + lane) * 3;   // subtile 1
    r1[0] = m1s; r1[1] = Z1; r1[2] = N1;
  }
  __syncthreads();
  // Wave 0: all 32 lanes reduce across waves (lane = ci*16 + col).
  if (wave == 0) {
    float mm = red[lane * 3 + 0], Zt = red[lane * 3 + 1], Nt = red[lane * 3 + 2];
    for (int w2 = 1; w2 < 8; ++w2) {
      float* r = red + (w2 * 32 + lane) * 3;
      float m2 = r[0], Z2 = r[1], N2 = r[2];
      float nm = fmaxf(mm, m2);
      float e1 = __expf(mm - nm), e2 = __expf(m2 - nm);
      Zt = Zt * e1 + Z2 * e2;
      Nt = Nt * e1 + N2 * e2;
      mm = nm;
    }
    int c = nb0 * 16 + lane;
    if (c < CC) logits[(size_t)b * CC + c] = Nt / Zt + out_b[c];
  }
}

__global__ void k_finalize(const float* __restrict__ acc, float* __restrict__ out) {
  if (threadIdx.x == 0 && blockIdx.x == 0) {
    float mse = acc[0] / (float)((long)ROWS * DD);
    float l1  = acc[1] / (float)((long)ROWS * MM);
    out[(size_t)BB * CC] = mse + 1.0e-3f * l1;
  }
}

// ---------------------------------------------------------------------------
extern "C" void kernel_launch(void* const* d_in, const int* in_sizes, int n_in,
                              void* d_out, int out_size, void* d_ws, size_t ws_size,
                              hipStream_t stream) {
  const float* x     = (const float*)d_in[0];
  const float* w_enc = (const float*)d_in[1];
  const float* b_enc = (const float*)d_in[2];
  const float* w_dec = (const float*)d_in[3];
  const float* b_dec = (const float*)d_in[4];
  const float* icd   = (const float*)d_in[5];
  const float* out_w = (const float*)d_in[6];
  const float* out_b = (const float*)d_in[7];
  float* out = (float*)d_out;

  char* ws = (char*)d_ws;
  size_t off = 0;
  float* acc = (float*)(ws + off);                       off += 256;
  unsigned short* x_bf    = (unsigned short*)(ws + off); off += (size_t)ROWS * DD * 2;
  unsigned short* wenc_pk = (unsigned short*)(ws + off); off += (size_t)KT_ENC * NT_ENC * 1024;
  unsigned short* wdec_pk = (unsigned short*)(ws + off); off += (size_t)KT_DEC * NT_DEC * 1024;
  unsigned short* icd_pk  = (unsigned short*)(ws + off); off += (size_t)KT_DEC * NT_C * 1024;
  unsigned short* ow_pk   = (unsigned short*)(ws + off); off += (size_t)KT_ENC * NT_C * 1024;
  unsigned short* f_bf    = (unsigned short*)(ws + off); off += (size_t)ROWS * MM * 2;
  (void)ws_size; (void)in_sizes; (void)n_in; (void)out_size;

  hipMemsetAsync(acc, 0, 256, stream);

  // bf16 conversion of activations
  k_f32_to_bf16<<<4096, 256, 0, stream>>>(x, x_bf, ROWS * DD);

  // Pack weight operands into WMMA-B tile layout (C padded to 8928, zero fill)
  k_pack_b<<<(KT_ENC * NT_ENC * 32 + 255) / 256, 256, 0, stream>>>(
      w_enc, wenc_pk, KT_ENC, NT_ENC, MM, (long)MM, 1L);           // (D,M) row-major
  k_pack_b<<<(KT_DEC * NT_DEC * 32 + 255) / 256, 256, 0, stream>>>(
      w_dec, wdec_pk, KT_DEC, NT_DEC, DD, (long)DD, 1L);           // (M,D) row-major
  k_pack_b<<<(KT_DEC * NT_C * 32 + 255) / 256, 256, 0, stream>>>(
      icd, icd_pk, KT_DEC, NT_C, CC, 1L, (long)MM);                // icd^T : (M,C)
  k_pack_b<<<(KT_ENC * NT_C * 32 + 255) / 256, 256, 0, stream>>>(
      out_w, ow_pk, KT_ENC, NT_C, CC, 1L, (long)DD);               // out_w^T : (D,C)

  // Encoder GEMM + ReLU + L1 reduction (2-way N blocking)
  k_encode<<<dim3(ROWS / 16, NT_ENC / 16), 256, 0, stream>>>(x_bf, wenc_pk, b_enc, f_bf, acc);

  // Decoder GEMM + MSE reduction (x_hat never materialized, 2-way N blocking)
  k_decode<<<dim3(ROWS / 16, NT_DEC / 16), 256, 0, stream>>>(f_bf, wdec_pk, b_dec, x, acc);

  // Fused dual-GEMM + online-softmax label attention (2x2 register blocking)
  size_t smem = (size_t)2 * (KT_DEC + KT_ENC) * 512 * sizeof(unsigned short)
              + 8 * 32 * 3 * sizeof(float);   // ~243 KB, within 320 KB WGP LDS
  hipFuncSetAttribute((const void*)k_label_attn,
                      hipFuncAttributeMaxDynamicSharedMemorySize, (int)smem);
  k_label_attn<<<dim3(CPAD / 32, BB), 256, smem, stream>>>(f_bf, x_bf, icd_pk, ow_pk, out_b, out);

  k_finalize<<<1, 1, 0, stream>>>(acc, out);
}
